// LMUFFT_601295421696
// MI455X (gfx1250) — matrix operand order repaired and checked by reading, not compiled, not run on hardware
//
#include <hip/hip_runtime.h>
#include <hip/hip_bf16.h>

typedef _Float16 half_t;
typedef __attribute__((ext_vector_type(16))) _Float16 v16h;
typedef __attribute__((ext_vector_type(8)))  _Float16 v8h;
typedef __attribute__((ext_vector_type(8)))  float    v8f;
typedef __attribute__((ext_vector_type(8)))  unsigned int v8u;

#define BATCH   32
#define IN_LEN  4096
#define IN_DIM  256
#define MEM     256
#define HIDDEN  512
#define CHAN    (MEM + IN_DIM)   // 512

// ---------------------------------------------------------------------------
// Generic f32 -> f16 conversion (grid-stride)
// ---------------------------------------------------------------------------
__global__ __launch_bounds__(256) void cvt_f32_to_f16(const float* __restrict__ src,
                                                      half_t* __restrict__ dst, int n) {
    for (int i = blockIdx.x * blockDim.x + threadIdx.x; i < n; i += gridDim.x * blockDim.x)
        dst[i] = (half_t)src[i];
}

// ---------------------------------------------------------------------------
// u[b,t] = relu( sum_i x[b,t,i] * Wu[i] ), stored as f16.
// One wave (32 lanes) per (b,t) row: each lane handles 8 contiguous floats.
// ---------------------------------------------------------------------------
__global__ __launch_bounds__(256) void compute_u(const float* __restrict__ x,
                                                 const float* __restrict__ Wu,
                                                 half_t* __restrict__ uh) {
    int gtid = blockIdx.x * blockDim.x + threadIdx.x;
    int wid  = gtid >> 5;            // global wave id == b*IN_LEN + t
    int lane = gtid & 31;
    if (wid >= BATCH * IN_LEN) return;
    const float4* row = (const float4*)(x + (size_t)wid * IN_DIM) + lane * 2;
    const float4* wv  = (const float4*)Wu + lane * 2;
    float4 a0 = row[0], a1 = row[1];
    float4 w0 = wv[0],  w1 = wv[1];
    float s = a0.x*w0.x + a0.y*w0.y + a0.z*w0.z + a0.w*w0.w
            + a1.x*w1.x + a1.y*w1.y + a1.z*w1.z + a1.w*w1.w;
    #pragma unroll
    for (int off = 16; off > 0; off >>= 1) s += __shfl_down(s, off, 32);
    if (lane == 0) uh[wid] = (half_t)fmaxf(s, 0.0f);
}

// ---------------------------------------------------------------------------
// Fill the x-half of the fused [m | x] f16 activation buffer.
// ---------------------------------------------------------------------------
__global__ __launch_bounds__(256) void fill_inph_x(const float* __restrict__ x,
                                                   half_t* __restrict__ inph) {
    const int total = BATCH * IN_LEN * IN_DIM;
    for (int idx = blockIdx.x * blockDim.x + threadIdx.x; idx < total;
         idx += gridDim.x * blockDim.x) {
        int row = idx >> 8;          // b*T + t
        int i   = idx & 255;
        inph[(size_t)row * CHAN + MEM + i] = (half_t)x[idx];
    }
}

// ---------------------------------------------------------------------------
// Causal convolution as Toeplitz matmul via WMMA f16.
//   m[b,t,k] = sum_{tau<=t} u[b, t-tau] * H[k, tau]
// Reversed u slab is stored in LDS as TWO parity-shifted packed-u32 copies so
// each lane's A fragment is 8 aligned ds_load_b32 (u32 packing == A-layout
// VGPR packing: K-even in [15:0], K-odd in [31:16]). Zero pad past Tend
// implements the causal mask on the diagonal chunks.
// ---------------------------------------------------------------------------
__global__ __launch_bounds__(256) void conv_wmma(const half_t* __restrict__ uh,
                                                 const half_t* __restrict__ Hh,
                                                 half_t* __restrict__ inph) {
    __shared__ unsigned int ur32[2][(IN_LEN + 64) / 2];   // 2 x 2080 u32

    const int b      = blockIdx.x >> 6;        // 64 slabs per batch
    const int tblock = (blockIdx.x & 63) * 64;
    const int Tend   = tblock + 63;
    const size_t ub  = (size_t)b * IN_LEN;

    // copy0 word q = halves (rev[2q], rev[2q+1]); copy1 word q = (rev[2q+1], rev[2q+2])
    // where rev[p] = u[b, Tend - p] for p <= Tend, else 0.
    const int nwords = (Tend + 51) / 2;
    for (int q = threadIdx.x; q < nwords; q += 256) {
        const int p0 = 2 * q, p1 = 2 * q + 1, p2 = 2 * q + 2;
        unsigned short s0 = (p0 <= Tend) ? __builtin_bit_cast(unsigned short, uh[ub + Tend - p0]) : (unsigned short)0;
        unsigned short s1 = (p1 <= Tend) ? __builtin_bit_cast(unsigned short, uh[ub + Tend - p1]) : (unsigned short)0;
        unsigned short s2 = (p2 <= Tend) ? __builtin_bit_cast(unsigned short, uh[ub + Tend - p2]) : (unsigned short)0;
        ur32[0][q] = ((unsigned)s1 << 16) | s0;
        ur32[1][q] = ((unsigned)s2 << 16) | s1;
    }
    __syncthreads();

    const int wave = threadIdx.x >> 5;
    const int lane = threadIdx.x & 31;
    const int n    = lane & 15;
    const bool hi  = lane >= 16;
    const int k0   = wave * 32;                // this wave's first k-tile

    v8f acc[4][2];
    #pragma unroll
    for (int it = 0; it < 4; ++it)
        #pragma unroll
        for (int kt = 0; kt < 2; ++kt)
            acc[it][kt] = (v8f)0.0f;

    // A[M,K] = rev[Tend - tt - M + c + K]; per-lane K runs {0..15}(+16 if hi)
    auto gatherA = [&](int tt, int c) -> v16h {
        const int base = Tend - tt - n + c + (hi ? 16 : 0);
        const int par  = base & 1;
        const int q0   = base >> 1;
        v8u aw;
        #pragma unroll
        for (int r = 0; r < 4; ++r) {
            aw[r]     = ur32[par][q0 + r];          // halves base+2r, base+2r+1
            aw[r + 4] = ur32[par][q0 + 4 + r];      // halves base+8+2r, +9
        }
        return __builtin_bit_cast(v16h, aw);
    };

    const int tb_off = hi ? 16 : 0;
    const int cmax   = tblock + 63;
    int c = 0;
    // Dense region: no causal masking possible, no skipped tiles.
    for (; c + 32 <= tblock; c += 32) {
        v16h bfr[2];
        #pragma unroll
        for (int kt = 0; kt < 2; ++kt) {
            const half_t* hp = Hh + (size_t)(k0 + kt * 16 + n) * IN_LEN + c + tb_off;
            bfr[kt] = *(const v16h*)hp;             // 32B aligned
            __builtin_prefetch(hp + 32, 0, 1);      // next chunk's B row
        }
        #pragma unroll
        for (int it = 0; it < 4; ++it) {
            v16h afr = gatherA(tblock + it * 16, c);
            #pragma unroll
            for (int kt = 0; kt < 2; ++kt)
                acc[it][kt] = __builtin_amdgcn_wmma_f32_16x16x32_f16(
                    false, afr, false, bfr[kt], (short)0, acc[it][kt], false, false);
        }
    }
    // Diagonal tail: 2 chunks with per-tile skip + zero-pad masking.
    for (; c <= cmax; c += 32) {
        v16h bfr[2];
        #pragma unroll
        for (int kt = 0; kt < 2; ++kt)
            bfr[kt] = *(const v16h*)(Hh + (size_t)(k0 + kt * 16 + n) * IN_LEN + c + tb_off);
        #pragma unroll
        for (int it = 0; it < 4; ++it) {
            const int tt = tblock + it * 16;
            if (c > tt + 15) continue;              // uniform branch: EXEC stays full
            v16h afr = gatherA(tt, c);
            #pragma unroll
            for (int kt = 0; kt < 2; ++kt)
                acc[it][kt] = __builtin_amdgcn_wmma_f32_16x16x32_f16(
                    false, afr, false, bfr[kt], (short)0, acc[it][kt], false, false);
        }
    }

    // Store m (f16). C layout: VGPR r -> M = r + (hi?8:0), N = n.
    #pragma unroll
    for (int it = 0; it < 4; ++it)
        #pragma unroll
        for (int kt = 0; kt < 2; ++kt)
            #pragma unroll
            for (int r = 0; r < 8; ++r) {
                const int t = tblock + it * 16 + r + (hi ? 8 : 0);
                const int k = k0 + kt * 16 + n;
                inph[(ub + t) * CHAN + k] = (half_t)acc[it][kt][r];
            }
}

// ---------------------------------------------------------------------------
// h = relu( [m|x] @ W_h^T ) via WMMA f16, f32 accumulate.
// Block: 64 rows x 512 outputs. The 64x512 f16 A tile is staged into LDS once
// per block with async-to-LDS DMA (ASYNCcnt), then every wave gathers aligned
// ds_load_b128 fragments. Wave w owns 4 o-tiles at o = 64*w. Taps t==4095
// rows into h_n.
// ---------------------------------------------------------------------------
__global__ __launch_bounds__(256) void gemm_wmma(const half_t* __restrict__ inph,
                                                 const half_t* __restrict__ Whh,
                                                 float* __restrict__ hout,
                                                 float* __restrict__ hn) {
    __shared__ half_t atile[64 * CHAN];            // 64 KB

    const size_t row0 = (size_t)blockIdx.x * 64;
    const int wave = threadIdx.x >> 5;
    const int lane = threadIdx.x & 31;
    const int n    = lane & 15;
    const bool hi  = lane >= 16;
    const int ow   = wave * 64;                    // 4 o-tiles per wave

    // Stage A tile: 64*512 halves, 128 per thread, via async DMA to LDS.
    {
        const half_t* gsrc = inph + row0 * CHAN;
        #pragma unroll
        for (int i = 0; i < 16; ++i) {
            const int idx = (threadIdx.x + i * 256) * 8;
            unsigned lds_off = (unsigned)(size_t)(atile + idx);
            unsigned long long gaddr = (unsigned long long)(size_t)(gsrc + idx);
            asm volatile("global_load_async_to_lds_b128 %0, %1, off"
                         :: "v"(lds_off), "v"(gaddr) : "memory");
        }
        asm volatile("s_wait_asynccnt 0x0" ::: "memory");
    }
    __syncthreads();

    v8f acc[4][4];
    #pragma unroll
    for (int it = 0; it < 4; ++it)
        #pragma unroll
        for (int kt = 0; kt < 4; ++kt)
            acc[it][kt] = (v8f)0.0f;

    for (int cc = 0; cc < CHAN; cc += 32) {
        const int cb = cc + (hi ? 16 : 0);
        v16h bfr[4];
        #pragma unroll
        for (int kt = 0; kt < 4; ++kt) {
            const half_t* wp = Whh + (size_t)(ow + kt * 16 + n) * CHAN + cb;
            bfr[kt] = *(const v16h*)wp;            // 32B aligned
            __builtin_prefetch(wp + 32, 0, 1);
        }
        #pragma unroll
        for (int it = 0; it < 4; ++it) {
            // A[M,K]: lane holds halves [c'..c'+15]
            const half_t* ap = atile + (it * 16 + n) * CHAN + cc + (hi ? 16 : 0);
            v8h lo  = *(const v8h*)ap;             // 16B-aligned ds_load_b128
            v8h hip = *(const v8h*)(ap + 8);
            v16h afr = __builtin_shufflevector(lo, hip,
                0, 1, 2, 3, 4, 5, 6, 7, 8, 9, 10, 11, 12, 13, 14, 15);
            #pragma unroll
            for (int kt = 0; kt < 4; ++kt)
                acc[it][kt] = __builtin_amdgcn_wmma_f32_16x16x32_f16(
                    false, afr, false, bfr[kt], (short)0, acc[it][kt], false, false);
        }
    }

    #pragma unroll
    for (int it = 0; it < 4; ++it)
        #pragma unroll
        for (int kt = 0; kt < 4; ++kt)
            #pragma unroll
            for (int r = 0; r < 8; ++r) {
                const size_t row = row0 + it * 16 + r + (hi ? 8 : 0);
                const int o = ow + kt * 16 + n;
                const float v = fmaxf(acc[it][kt][r], 0.0f);
                hout[row * HIDDEN + o] = v;
                if ((row & (IN_LEN - 1)) == (IN_LEN - 1))
                    hn[(row >> 12) * HIDDEN + o] = v;   // row>>12 == batch index
            }
}

// ---------------------------------------------------------------------------
// Launcher. Inputs: x(f32), W_u(f32), W_h(f32), H(f32). Output: [h | h_n] f32.
// ---------------------------------------------------------------------------
extern "C" void kernel_launch(void* const* d_in, const int* in_sizes, int n_in,
                              void* d_out, int out_size, void* d_ws, size_t ws_size,
                              hipStream_t stream) {
    const float* x  = (const float*)d_in[0];
    const float* Wu = (const float*)d_in[1];
    const float* Wh = (const float*)d_in[2];
    const float* H  = (const float*)d_in[3];

    float* hout = (float*)d_out;
    float* hn   = hout + (size_t)BATCH * IN_LEN * HIDDEN;

    // Workspace layout (f16 buffers, 256B-aligned offsets)
    char* ws = (char*)d_ws;
    half_t* uh   = (half_t*)(ws);                               // 256 KB
    half_t* Hh   = (half_t*)(ws + (1u << 20));                  // 2 MB
    half_t* Whh  = (half_t*)(ws + (4u << 20));                  // 512 KB
    half_t* inph = (half_t*)(ws + (8u << 20));                  // 128 MB

    // 1) Precision staging
    cvt_f32_to_f16<<<4096, 256, 0, stream>>>(H,  Hh,  MEM * IN_LEN);
    cvt_f32_to_f16<<<1024, 256, 0, stream>>>(Wh, Whh, HIDDEN * CHAN);
    compute_u<<<(BATCH * IN_LEN * 32) / 256, 256, 0, stream>>>(x, Wu, uh);
    fill_inph_x<<<16384, 256, 0, stream>>>(x, inph);

    // 2) Causal convolution (Toeplitz x H^T) via WMMA -> low half of inph
    conv_wmma<<<BATCH * (IN_LEN / 64), 256, 0, stream>>>(uh, Hh, inph);

    // 3) Output GEMM with fused relu + h_n tap
    gemm_wmma<<<BATCH * IN_LEN / 64, 256, 0, stream>>>(inph, Whh, hout, hn);
}
